// TrajGRUCell_75247827026127
// MI455X (gfx1250) — matrix operand order repaired and checked by reading, not compiled, not run on hardware
//
#include <hip/hip_runtime.h>
#include <hip/hip_bf16.h>
#include <math.h>

// ---------------------------------------------------------------------------
// TrajGRU cell for MI455X (gfx1250, wave32). ALL conv/matmul work on WMMA.
// Tap-decomposed convs use ZERO-HALO padded activation images (68x68) so the
// shifted A-fragment loads are unconditional -> no EXEC-masked branches, no
// per-WMMA s_wait_loadcnt 0x0 drains; the 25-tap loops software-pipeline.
//
//  1) transpose_h : h -> hTp bf16 [b][68][68][128]       (zero halo)
//  2) im2col_x    : x -> xI bf16 [pixel][K=416]          (25 taps x 16ch, pad->416)
//  3) pack_*      : weights -> bf16 B operands [N][K]
//  4) gemm<416,32>: f_pre = xI x Wgx                     (x branch of flow gen)
//  5) conv_h      : f = leaky(f_pre + tap GEMM hTp x Wgh + bias) -> fbfp (halo'd)
//  6) conv_f      : flows = tap GEMM fbfp x Wgf + gf_b   (25 WMMAs/wave)
//  7) warp_bl     : bilinear gather of h along 5 flows -> warped bf16 [pixel][640]
//  8) gemm<640,256>: gates = warped x (whr|whh)          (z gate algebraically dead)
//  9) gemm<416,256>: xg    = xI x (wxr|wxh)
// 10) finalize    : r = sigmoid(xr+b + tr+sum(whr_b));
//                   h_new = leaky(xh+b + r*(th+sum(whh_b)));  ((1-z)ht+z·ht == ht)
// ---------------------------------------------------------------------------

typedef __attribute__((ext_vector_type(16))) __bf16       v16bf;
typedef __attribute__((ext_vector_type(8)))  float        v8f;
typedef __attribute__((ext_vector_type(4))) unsigned int  v4u;

#define Bn   16
#define CIN  16
#define CH   128
#define HW   4096                      // 64*64
#define LL   5
#define NPIX (Bn * HW)                 // 65536  (GEMM M)
#define KG   (LL * CH)                 // 640    gate-GEMM K
#define KX   416                       // x im2col K (400 used, zero padded)
#define NN   256                       // [0,128)=r gate, [128,256)=h gate
#define PW   68                        // padded width/height (2-px halo)
#define PHW  (PW * PW)                 // 4624 padded pixels per image
#define NEG_SLOPE 0.2f

// ---- workspace layout (bytes); REGION A and REGION G are reused -----------
#define REGA_OFF   ((size_t)0)                            // warped bf16 [NPIX][640] (84MB), later xg f32 [NPIX][256] (67MB)
#define WPACK_OFF  (REGA_OFF  + (size_t)NPIX * KG * 2)    // bf16 [256][640]
#define HT_OFF     (WPACK_OFF + (size_t)NN * KG * 2)      // bf16 [Bn][PHW][128] padded
#define XI_OFF     (HT_OFF    + (size_t)Bn * PHW * CH * 2)// bf16 [NPIX][416]
#define WGX_OFF    (XI_OFF    + (size_t)NPIX * KX * 2)    // bf16 [32][416]
#define WX2_OFF    (WGX_OFF   + (size_t)32  * KX * 2)     // bf16 [256][416]
#define WGH_OFF    (WX2_OFF   + (size_t)NN  * KX * 2)     // bf16 [25][32][128]
#define WGF_OFF    (WGH_OFF   + (size_t)25 * 32 * CH * 2) // bf16 [25][16][32]
#define FBF_OFF    (WGF_OFF   + (size_t)25 * 16 * 32 * 2) // bf16 [Bn][PHW][32] padded
#define FLOWS_OFF  (FBF_OFF   + (size_t)Bn * PHW * 32 * 2)// f32  [B][10][HW]
#define REGG_OFF   (FLOWS_OFF + (size_t)Bn * 10 * HW * 4) // f_pre f32 [NPIX][32], later gates f32 [NPIX][256]

__device__ __forceinline__ unsigned short f2bf(float f) {
  unsigned int u = __float_as_uint(f);
  u += 0x7fffu + ((u >> 16) & 1u);     // round-to-nearest-even
  return (unsigned short)(u >> 16);
}
__device__ __forceinline__ int iclamp(int v, int lo, int hi) {
  return v < lo ? lo : (v > hi ? hi : v);
}

// ---- 1) h [B,128,H,W] -> padded hTp bf16 [b][68][68][128], zero halo ------
__global__ void transpose_h(const float* __restrict__ h, unsigned short* __restrict__ hTp) {
  int t = blockIdx.x * blockDim.x + threadIdx.x;          // Bn*PHW*128 = 9,469,952
  int c = t & 127, pp = t >> 7;
  int pi = pp % PHW, b = pp / PHW;
  int xx = pi % PW - 2, yy = pi / PW - 2;
  unsigned short v = 0;
  if (xx >= 0 && xx < 64 && yy >= 0 && yy < 64)
    v = f2bf(h[((size_t)(b * CH + c)) * HW + yy * 64 + xx]);
  hTp[t] = v;
}

// ---- zero the padded f buffer (halo must be 0 before conv_h interior) -----
__global__ void zero_u16(unsigned short* __restrict__ p) {
  p[blockIdx.x * blockDim.x + threadIdx.x] = 0;
}

// ---- 2) x [B,16,H,W] -> im2col bf16 [pixel][416] (k = tap*16 + c) ---------
__global__ void im2col_x(const float* __restrict__ x, unsigned short* __restrict__ xI) {
  size_t t = (size_t)blockIdx.x * blockDim.x + threadIdx.x;  // NPIX*416
  int k = (int)(t % KX); size_t pix = t / KX;
  int xx = (int)(pix & 63), yy = (int)((pix >> 6) & 63), b = (int)(pix >> 12);
  unsigned short v = 0;
  if (k < 400) {
    int tap = k >> 4, c = k & 15;
    int sy = yy + (tap / 5) - 2, sx = xx + (tap % 5) - 2;
    if (sy >= 0 && sy < 64 && sx >= 0 && sx < 64)
      v = f2bf(x[((size_t)(b * CIN + c)) * HW + sy * 64 + sx]);
  }
  xI[t] = v;
}

// ---- 3a) pack 5x5 conv weights [N][16][5][5] -> [N][416] bf16 -------------
__global__ void pack_w5(const float* __restrict__ w, unsigned short* __restrict__ dst, int N) {
  int t = blockIdx.x * blockDim.x + threadIdx.x;
  if (t >= N * KX) return;
  int k = t % KX, n = t / KX;
  unsigned short v = 0;
  if (k < 400) v = f2bf(w[(n * 16 + (k & 15)) * 25 + (k >> 4)]);
  dst[t] = v;
}

// ---- 3b) pack gh_w [32][128][25] -> [tap][n=32][k=128] bf16 ---------------
__global__ void pack_gh(const float* __restrict__ w, unsigned short* __restrict__ dst) {
  int t = blockIdx.x * blockDim.x + threadIdx.x;          // 25*32*128
  if (t >= 25 * 32 * CH) return;
  int k = t & 127, n = (t >> 7) & 31, tap = t >> 12;
  dst[t] = f2bf(w[(n * CH + k) * 25 + tap]);
}

// ---- 3c) pack gf_w [10][32][25] -> [tap][n=16][k=32] bf16 (n>=10 zero) ----
__global__ void pack_gf(const float* __restrict__ w, unsigned short* __restrict__ dst) {
  int t = blockIdx.x * blockDim.x + threadIdx.x;          // 25*16*32
  if (t >= 25 * 16 * 32) return;
  int k = t & 31, n = (t >> 5) & 15, tap = t >> 9;
  dst[t] = (n < 10) ? f2bf(w[(n * 32 + k) * 25 + tap]) : (unsigned short)0;
}

// ---- 3d) pack whr/whh [L][128][128] -> gate B operand [256][640] ----------
__global__ void pack_gate(const float* __restrict__ whr_w, const float* __restrict__ whh_w,
                          unsigned short* __restrict__ wpack) {
  int t = blockIdx.x * blockDim.x + threadIdx.x;          // 256*640
  if (t >= NN * KG) return;
  int k = t % KG, n = t / KG;
  int l = k / CH, c = k % CH, o = n & (CH - 1);
  const float* w = (n < CH) ? whr_w : whh_w;
  wpack[t] = f2bf(w[(l * CH + o) * CH + c]);
}

// ---- generic WMMA GEMM: C[M][LDC] = A[M][KDIM] x B[N][KDIM]^T -------------
template<int KDIM, int LDC>
__global__ void __launch_bounds__(32)
gemm_bf16nt(const unsigned short* __restrict__ A, const unsigned short* __restrict__ Bw,
            float* __restrict__ C) {
  const int lane = threadIdx.x, row = lane & 15, hi = lane >> 4;
  const int m0 = blockIdx.x << 4, n0 = blockIdx.y << 4;
  const unsigned short* arow = A  + (size_t)(m0 + row) * KDIM + hi * 8;
  const unsigned short* brow = Bw + (size_t)(n0 + row) * KDIM + hi * 8;
  v8f acc = {};
  union Frag { v4u u[2]; v16bf v; };
#pragma unroll
  for (int kt = 0; kt < KDIM / 32; ++kt) {
    Frag af, bf_;
    af.u[0]  = *(const v4u*)(arow + kt * 32);        // A 16x32: K hi*8+0..7
    af.u[1]  = *(const v4u*)(arow + kt * 32 + 16);   //          K 16+hi*8+0..7
    bf_.u[0] = *(const v4u*)(brow + kt * 32);        // B 32x16: K hi*8+0..7
    bf_.u[1] = *(const v4u*)(brow + kt * 32 + 16);   //          K 16+hi*8+0..7
    if (kt + 1 < KDIM / 32) {
      __builtin_prefetch(arow + (kt + 1) * 32, 0, 1);
      __builtin_prefetch(brow + (kt + 1) * 32, 0, 1);
    }
    acc = __builtin_amdgcn_wmma_f32_16x16x32_bf16(false, af.v, false, bf_.v,
                                                  (short)0, acc, false, false);
  }
  float* c = C + (size_t)m0 * LDC + n0 + row;
#pragma unroll
  for (int r = 0; r < 8; ++r)                        // D: M = r + 8*hi, N = row
    c[(size_t)(r + hi * 8) * LDC] = acc[r];
}

// ---- 5) tap-decomposed h-conv + x branch + bias + leaky -> padded f bf16 --
// 25 taps x 4 K-tiles = 100 unconditional WMMAs per wave (zero halo).
__global__ void __launch_bounds__(32)
conv_h(const unsigned short* __restrict__ hTp,  // [Bn][PHW][128] padded
       const unsigned short* __restrict__ Wgh,  // [25][32][128]
       const float* __restrict__ fpre,          // [NPIX][32]
       const float* __restrict__ gxb, const float* __restrict__ ghb,
       unsigned short* __restrict__ fbfp) {     // [Bn][PHW][32] padded
  const int lane = threadIdx.x, row = lane & 15, hi = lane >> 4;
  const int m0 = blockIdx.x << 4, n0 = blockIdx.y << 4;
  const int xx = (m0 & 63) + row, yy = (m0 >> 6) & 63, b = m0 >> 12;
  v8f acc = {};
  union Frag { v4u u[2]; v16bf v; };
#pragma unroll
  for (int ky = 0; ky < 5; ++ky) {
#pragma unroll
    for (int kx = 0; kx < 5; ++kx) {
      // padded coords: (yy+ky-2)+2 = yy+ky, (xx+kx-2)+2 = xx+kx  (always in-bounds)
      const unsigned short* arow =
          hTp + ((size_t)(b * PHW + (yy + ky) * PW + (xx + kx))) * CH + hi * 8;
      const unsigned short* brow =
          Wgh + ((size_t)((ky * 5 + kx) * 32 + n0 + row)) * CH + hi * 8;
#pragma unroll
      for (int kt = 0; kt < CH / 32; ++kt) {
        Frag af, bf_;
        af.u[0]  = *(const v4u*)(arow + kt * 32);
        af.u[1]  = *(const v4u*)(arow + kt * 32 + 16);
        bf_.u[0] = *(const v4u*)(brow + kt * 32);
        bf_.u[1] = *(const v4u*)(brow + kt * 32 + 16);
        acc = __builtin_amdgcn_wmma_f32_16x16x32_bf16(false, af.v, false, bf_.v,
                                                      (short)0, acc, false, false);
      }
    }
  }
  int n = n0 + row;
  float bias = gxb[n] + ghb[n];
#pragma unroll
  for (int r = 0; r < 8; ++r) {
    int m = m0 + r + hi * 8;
    int mx = m & 63, my = (m >> 6) & 63, mb = m >> 12;
    float v = acc[r] + fpre[(size_t)m * 32 + n] + bias;
    v = v >= 0.f ? v : NEG_SLOPE * v;
    fbfp[((size_t)(mb * PHW + (my + 2) * PW + (mx + 2))) * 32 + n] = f2bf(v);
  }
}

// ---- 6) tap-decomposed f-conv -> flows (25 unconditional WMMAs/wave) ------
__global__ void __launch_bounds__(32)
conv_f(const unsigned short* __restrict__ fbfp, // [Bn][PHW][32] padded
       const unsigned short* __restrict__ Wgf,  // [25][16][32]
       const float* __restrict__ gfb,
       float* __restrict__ flows) {             // [B][10][HW]
  const int lane = threadIdx.x, row = lane & 15, hi = lane >> 4;
  const int m0 = blockIdx.x << 4;
  const int xx = (m0 & 63) + row, yy = (m0 >> 6) & 63, b = m0 >> 12;
  v8f acc = {};
  union Frag { v4u u[2]; v16bf v; };
#pragma unroll
  for (int ky = 0; ky < 5; ++ky) {
#pragma unroll
    for (int kx = 0; kx < 5; ++kx) {
      const unsigned short* arow =
          fbfp + ((size_t)(b * PHW + (yy + ky) * PW + (xx + kx))) * 32 + hi * 8;
      const unsigned short* brow =
          Wgf + ((size_t)((ky * 5 + kx) * 16 + row)) * 32 + hi * 8;
      Frag af, bf_;
      af.u[0]  = *(const v4u*)(arow);
      af.u[1]  = *(const v4u*)(arow + 16);
      bf_.u[0] = *(const v4u*)(brow);
      bf_.u[1] = *(const v4u*)(brow + 16);
      acc = __builtin_amdgcn_wmma_f32_16x16x32_bf16(false, af.v, false, bf_.v,
                                                    (short)0, acc, false, false);
    }
  }
  int n = row;                                   // flow channel (0..9 valid)
  if (n < 10) {
#pragma unroll
    for (int r = 0; r < 8; ++r) {
      int m = m0 + r + hi * 8;
      int sp = m & (HW - 1), bb = m >> 12;
      flows[((size_t)(bb * 10 + n)) * HW + sp] = acc[r] + gfb[n];
    }
  }
}

// ---- 7) bilinear warp -> bf16 A operand [pixel][l*128+c] ------------------
__global__ void warp_bl(const float* __restrict__ h, const float* __restrict__ flows,
                        unsigned short* __restrict__ warped) {
  int t = blockIdx.x * blockDim.x + threadIdx.x;          // B*L*HW = 327,680
  if (t >= Bn * LL * HW) return;
  int xx = t & 63, yy = (t >> 6) & 63, rem = t >> 12;
  int l = rem % LL, b = rem / LL;
  int sp = yy * 64 + xx;
  float px = flows[((size_t)(b * 10 + 2 * l)) * HW + sp] + (float)xx;
  float py = flows[((size_t)(b * 10 + 2 * l + 1)) * HW + sp] + (float)yy;
  float x0f = floorf(px), y0f = floorf(py);
  int x0 = (int)x0f, y0 = (int)y0f;
  float ax = px - x0f, ay = py - y0f;
  float w00 = (1.f - ax) * (1.f - ay), w10 = ax * (1.f - ay);
  float w01 = (1.f - ax) * ay,         w11 = ax * ay;
  bool bx0 = (x0 >= 0) & (x0 <= 63), bx1 = (x0 + 1 >= 0) & (x0 + 1 <= 63);
  bool by0 = (y0 >= 0) & (y0 <= 63), by1 = (y0 + 1 >= 0) & (y0 + 1 <= 63);
  if (!(bx0 & by0)) w00 = 0.f;
  if (!(bx1 & by0)) w10 = 0.f;
  if (!(bx0 & by1)) w01 = 0.f;
  if (!(bx1 & by1)) w11 = 0.f;
  int x0c = iclamp(x0, 0, 63), x1c = iclamp(x0 + 1, 0, 63);
  int y0c = iclamp(y0, 0, 63), y1c = iclamp(y0 + 1, 0, 63);
  int i00 = y0c * 64 + x0c, i10 = y0c * 64 + x1c, i01 = y1c * 64 + x0c, i11 = y1c * 64 + x1c;
  const float* hb = h + (size_t)b * CH * HW;
  unsigned short* dst = warped + ((size_t)(b * HW + sp)) * KG + l * CH;
  for (int c = 0; c < CH; ++c) {
    const float* hc = hb + (size_t)c * HW;
    float v = w00 * hc[i00] + w10 * hc[i10] + w01 * hc[i01] + w11 * hc[i11];
    dst[c] = f2bf(v);
  }
}

// ---- 10) gate math; h_new = h_temp (z path algebraically cancels) ---------
__global__ void finalize(const float* __restrict__ gates,  // [pixel][256] (tr|th)
                         const float* __restrict__ xg,     // [pixel][256] (xr|xh)
                         const float* __restrict__ wxr_b, const float* __restrict__ wxh_b,
                         const float* __restrict__ whr_b, const float* __restrict__ whh_b,
                         float* __restrict__ out) {
  int t = blockIdx.x * blockDim.x + threadIdx.x;           // B*128*HW = 2^23
  int xx = t & 63, yy = (t >> 6) & 63, o = (t >> 12) & 127, b = t >> 19;
  size_t pix = (size_t)b * HW + yy * 64 + xx;
  float tr = gates[pix * NN + o];
  float th = gates[pix * NN + 128 + o];
  float br = 0.f, bh = 0.f;
#pragma unroll
  for (int l = 0; l < LL; ++l) { br += whr_b[l * CH + o]; bh += whh_b[l * CH + o]; }
  float xr = xg[pix * NN + o]       + wxr_b[o];
  float xh = xg[pix * NN + 128 + o] + wxh_b[o];
  float r  = 1.f / (1.f + __expf(-(xr + tr + br)));
  float ht = xh + r * (th + bh);
  ht = ht >= 0.f ? ht : NEG_SLOPE * ht;
  out[t] = ht;
  out[(size_t)(1u << 23) + t] = ht;                        // tuple (h_new, h_new)
}

// ---------------------------------------------------------------------------
extern "C" void kernel_launch(void* const* d_in, const int* in_sizes, int n_in,
                              void* d_out, int out_size, void* d_ws, size_t ws_size,
                              hipStream_t stream) {
  (void)in_sizes; (void)n_in; (void)out_size; (void)ws_size;
  const float* x     = (const float*)d_in[0];
  const float* h     = (const float*)d_in[1];
  const float* gx_w  = (const float*)d_in[2];
  const float* gx_b  = (const float*)d_in[3];
  const float* gh_w  = (const float*)d_in[4];
  const float* gh_b  = (const float*)d_in[5];
  const float* gf_w  = (const float*)d_in[6];
  const float* gf_b  = (const float*)d_in[7];
  // d_in[8..9] = wxz_w/wxz_b, d_in[14..15] = whz_w/whz_b: algebraically dead
  const float* wxr_w = (const float*)d_in[10];
  const float* wxr_b = (const float*)d_in[11];
  const float* wxh_w = (const float*)d_in[12];
  const float* wxh_b = (const float*)d_in[13];
  const float* whr_w = (const float*)d_in[16];
  const float* whr_b = (const float*)d_in[17];
  const float* whh_w = (const float*)d_in[18];
  const float* whh_b = (const float*)d_in[19];

  char* ws = (char*)d_ws;
  unsigned short* warped = (unsigned short*)(ws + REGA_OFF);   // then dead
  float*          xgbuf  = (float*)         (ws + REGA_OFF);   // reuses REGION A
  unsigned short* wpack  = (unsigned short*)(ws + WPACK_OFF);
  unsigned short* hTp    = (unsigned short*)(ws + HT_OFF);
  unsigned short* xI     = (unsigned short*)(ws + XI_OFF);
  unsigned short* Wgx    = (unsigned short*)(ws + WGX_OFF);
  unsigned short* Wx2    = (unsigned short*)(ws + WX2_OFF);
  unsigned short* Wgh    = (unsigned short*)(ws + WGH_OFF);
  unsigned short* Wgf    = (unsigned short*)(ws + WGF_OFF);
  unsigned short* fbfp   = (unsigned short*)(ws + FBF_OFF);
  float*          flows  = (float*)         (ws + FLOWS_OFF);
  float*          fpre   = (float*)         (ws + REGG_OFF);   // then dead
  float*          gates  = (float*)         (ws + REGG_OFF);   // reuses REGION G
  float*          out    = (float*)d_out;

  transpose_h<<<(Bn * PHW * CH) / 256, 256, 0, stream>>>(h, hTp);
  zero_u16<<<(Bn * PHW * 32) / 256, 256, 0, stream>>>(fbfp);
  im2col_x<<<((size_t)NPIX * KX) / 256, 256, 0, stream>>>(x, xI);
  pack_w5<<<(32 * KX + 255) / 256, 256, 0, stream>>>(gx_w, Wgx, 32);
  pack_w5<<<(128 * KX + 255) / 256, 256, 0, stream>>>(wxr_w, Wx2, 128);
  pack_w5<<<(128 * KX + 255) / 256, 256, 0, stream>>>(wxh_w, Wx2 + (size_t)128 * KX, 128);
  pack_gh<<<(25 * 32 * CH + 255) / 256, 256, 0, stream>>>(gh_w, Wgh);
  pack_gf<<<(25 * 16 * 32 + 255) / 256, 256, 0, stream>>>(gf_w, Wgf);
  pack_gate<<<(NN * KG + 255) / 256, 256, 0, stream>>>(whr_w, whh_w, wpack);

  gemm_bf16nt<KX, 32><<<dim3(NPIX / 16, 2), 32, 0, stream>>>(xI, Wgx, fpre);
  conv_h<<<dim3(NPIX / 16, 2), 32, 0, stream>>>(hTp, Wgh, fpre, gx_b, gh_b, fbfp);
  conv_f<<<dim3(NPIX / 16, 1), 32, 0, stream>>>(fbfp, Wgf, gf_b, flows);
  warp_bl<<<(Bn * LL * HW + 255) / 256, 256, 0, stream>>>(h, flows, warped);
  gemm_bf16nt<KG, NN><<<dim3(NPIX / 16, NN / 16), 32, 0, stream>>>(warped, wpack, gates);
  gemm_bf16nt<KX, NN><<<dim3(NPIX / 16, NN / 16), 32, 0, stream>>>(xI, Wx2, xgbuf);
  finalize<<<(Bn * CH * HW) / 256, 256, 0, stream>>>(gates, xgbuf, wxr_b, wxh_b,
                                                     whr_b, whh_b, out);
}
